// PointNetSetAbstraction_68186900791886
// MI455X (gfx1250) — matrix operand order repaired and checked by reading, not compiled, not run on hardware
//
#include <hip/hip_runtime.h>

typedef __attribute__((ext_vector_type(16))) _Float16 v16h;
typedef __attribute__((ext_vector_type(8)))  _Float16 v8h;
typedef __attribute__((ext_vector_type(8)))  float    v8f;

#define NPTS    4096
#define NPOINT_ 1024
#define R2      0.04f

static __device__ inline v16h cat8(v8h lo, v8h hi) {
    return __builtin_shufflevector(lo, hi, 0, 1, 2, 3, 4, 5, 6, 7,
                                           8, 9, 10, 11, 12, 13, 14, 15);
}

// ---------------------------------------------------------------------------
// Prep: f32 weights -> padded f16 weights; fold conv-bias + batchnorm into
// per-channel alpha/beta:  out = relu(alpha*conv + beta)
// ---------------------------------------------------------------------------
__global__ __launch_bounds__(256) void prep_kernel(
    const float* __restrict__ w0, const float* __restrict__ b0,
    const float* __restrict__ g0, const float* __restrict__ be0,
    const float* __restrict__ m0, const float* __restrict__ v0,
    const float* __restrict__ w1, const float* __restrict__ b1,
    const float* __restrict__ g1, const float* __restrict__ be1,
    const float* __restrict__ m1, const float* __restrict__ v1,
    const float* __restrict__ w2, const float* __restrict__ b2,
    const float* __restrict__ g2, const float* __restrict__ be2,
    const float* __restrict__ m2, const float* __restrict__ v2,
    _Float16* __restrict__ w0h, _Float16* __restrict__ w1h,
    _Float16* __restrict__ w2h, float* __restrict__ ab)
{
    int t = threadIdx.x;
    for (int i = t; i < 64 * 96; i += 256) {
        int o = i / 96, c = i % 96;
        w0h[i] = (c < 67) ? (_Float16)w0[o * 67 + c] : (_Float16)0.f;
    }
    for (int i = t; i < 64 * 64; i += 256)  w1h[i] = (_Float16)w1[i];
    for (int i = t; i < 128 * 64; i += 256) w2h[i] = (_Float16)w2[i];
    if (t < 64) {
        float s0 = g0[t] * rsqrtf(v0[t] + 1e-5f);
        ab[t]       = s0;
        ab[64 + t]  = s0 * (b0[t] - m0[t]) + be0[t];
        float s1 = g1[t] * rsqrtf(v1[t] + 1e-5f);
        ab[128 + t] = s1;
        ab[192 + t] = s1 * (b1[t] - m1[t]) + be1[t];
    }
    if (t < 128) {
        float s2 = g2[t] * rsqrtf(v2[t] + 1e-5f);
        ab[256 + t] = s2;
        ab[384 + t] = s2 * (b2[t] - m2[t]) + be2[t];
    }
}

// ---------------------------------------------------------------------------
// Farthest point sampling: one workgroup (512 thr = 16 waves) per batch.
// ---------------------------------------------------------------------------
__global__ __launch_bounds__(512) void fps_kernel(
    const float* __restrict__ xyz, int* __restrict__ fps_idx,
    float* __restrict__ new_xyz_ws, float* __restrict__ out_newxyz)
{
    const int b = blockIdx.x;
    const int t = threadIdx.x;
    const int lane = t & 31, wid = t >> 5;
    const float* xb = xyz + (size_t)b * 3 * NPTS;

    float px[8], py[8], pz[8], dist[8];
    #pragma unroll
    for (int i = 0; i < 8; ++i) {
        int n = t + 512 * i;
        px[i] = xb[n];
        py[i] = xb[NPTS + n];
        pz[i] = xb[2 * NPTS + n];
        dist[i] = 1e10f;
    }

    __shared__ float s_val[16];
    __shared__ int   s_idx[16];
    __shared__ int   s_far;

    int far = 0;
    for (int it = 0; it < NPOINT_; ++it) {
        // far is wave-uniform -> scalar loads of the centroid
        float cx = xb[far], cy = xb[NPTS + far], cz = xb[2 * NPTS + far];
        if (t == 0) {
            fps_idx[b * NPOINT_ + it] = far;
            new_xyz_ws[(b * NPOINT_ + it) * 3 + 0] = cx;
            new_xyz_ws[(b * NPOINT_ + it) * 3 + 1] = cy;
            new_xyz_ws[(b * NPOINT_ + it) * 3 + 2] = cz;
            out_newxyz[(b * 3 + 0) * NPOINT_ + it] = cx;
            out_newxyz[(b * 3 + 1) * NPOINT_ + it] = cy;
            out_newxyz[(b * 3 + 2) * NPOINT_ + it] = cz;
        }
        float best = -1.f; int bidx = 0;
        #pragma unroll
        for (int i = 0; i < 8; ++i) {
            float dx = px[i] - cx, dy = py[i] - cy, dz = pz[i] - cz;
            float d = dx * dx + dy * dy + dz * dz;
            dist[i] = fminf(dist[i], d);
            if (dist[i] > best) { best = dist[i]; bidx = t + 512 * i; }
        }
        // wave argmax (tie -> smallest index)
        #pragma unroll
        for (int off = 16; off >= 1; off >>= 1) {
            float ov = __shfl_down(best, off, 32);
            int   oi = __shfl_down(bidx, off, 32);
            if (ov > best || (ov == best && oi < bidx)) { best = ov; bidx = oi; }
        }
        if (lane == 0) { s_val[wid] = best; s_idx[wid] = bidx; }
        __syncthreads();
        if (t == 0) {
            float bv = s_val[0]; int bi = s_idx[0];
            for (int w = 1; w < 16; ++w)
                if (s_val[w] > bv || (s_val[w] == bv && s_idx[w] < bi)) { bv = s_val[w]; bi = s_idx[w]; }
            s_far = bi;
        }
        __syncthreads();
        far = __builtin_amdgcn_readfirstlane(s_far);
    }
}

// ---------------------------------------------------------------------------
// Ball query: one wave per centroid; ballot scan gives first-32 in index order.
// ---------------------------------------------------------------------------
__global__ __launch_bounds__(256) void ball_kernel(
    const float* __restrict__ xyz, const float* __restrict__ new_xyz_ws,
    int* __restrict__ ball_idx)
{
    const int t = threadIdx.x, lane = t & 31;
    const int wid = __builtin_amdgcn_readfirstlane(t >> 5);
    const int gid = blockIdx.x * 8 + wid;           // 0 .. 16383 (scalar)
    const int b = gid >> 10;
    const float* xb = xyz + (size_t)b * 3 * NPTS;
    const float cx = new_xyz_ws[gid * 3 + 0];
    const float cy = new_xyz_ws[gid * 3 + 1];
    const float cz = new_xyz_ws[gid * 3 + 2];

    int count = 0, firstN = 0;
    bool none = true;
    for (int base = 0; base < NPTS; base += 32) {
        int n = base + lane;
        float dx = xb[n] - cx, dy = xb[NPTS + n] - cy, dz = xb[2 * NPTS + n] - cz;
        bool in = (dx * dx + dy * dy + dz * dz) <= R2;
        unsigned mask = (unsigned)__ballot(in);
        if (mask && none) { firstN = base + __builtin_ctz(mask); none = false; }
        int rank = __popc(mask & ((1u << lane) - 1u));
        if (in && (count + rank) < 32) ball_idx[gid * 32 + count + rank] = n;
        count = __builtin_amdgcn_readfirstlane(count + __popc(mask));
        if (count >= 32) break;                      // scalar branch
    }
    if (count < 32 && lane >= count)
        ball_idx[gid * 32 + lane] = firstN;          // pad with first neighbor
}

// ---------------------------------------------------------------------------
// WMMA GEMM layer, fully unrolled.  Per wave: fixed n0 = wid*16, B fragments
// loaded once and reused across all M-tiles.  bn+relu epilogue, store f16
// back to LDS laid out [pos][channel].
// ---------------------------------------------------------------------------
template <int CPAD, int SRC_STRIDE, int DST_STRIDE, int COUT, int KDIM>
__device__ inline void gemm_layer(const _Float16* __restrict__ W,
                                  const _Float16* Hsrc, _Float16* Hdst,
                                  const float* __restrict__ alpha,
                                  const float* __restrict__ beta,
                                  int lane, int wid)
{
    constexpr int MT = COUT / 16;
    constexpr int KS = KDIM / 32;
    const int hipart = (lane & 16) >> 1;             // 0 or 8
    const int pos = (wid << 4) + (lane & 15);
    const _Float16* hcol = Hsrc + pos * SRC_STRIDE + hipart;

    v16h bfrag[KS];
    #pragma unroll
    for (int kk = 0; kk < KS; ++kk) {
        v8h blo = *(const v8h*)(hcol + kk * 32);
        v8h bhi = *(const v8h*)(hcol + kk * 32 + 16);
        bfrag[kk] = cat8(blo, bhi);
    }

    #pragma unroll
    for (int j = 0; j < MT; ++j) {
        const int m0 = j << 4;
        v8f acc = {0.f, 0.f, 0.f, 0.f, 0.f, 0.f, 0.f, 0.f};
        const _Float16* wrow = W + (m0 + (lane & 15)) * CPAD + hipart;
        #pragma unroll
        for (int kk = 0; kk < KS; ++kk) {
            v8h alo = *(const v8h*)(wrow + kk * 32);
            v8h ahi = *(const v8h*)(wrow + kk * 32 + 16);
            acc = __builtin_amdgcn_wmma_f32_16x16x32_f16(
                false, cat8(alo, ahi), false, bfrag[kk], (short)0, acc, false, false);
        }
        v8h outh;
        const int chbase = m0 + hipart;
        #pragma unroll
        for (int r = 0; r < 8; ++r) {
            float vv = acc[r] * alpha[chbase + r] + beta[chbase + r];
            outh[r] = (_Float16)fmaxf(vv, 0.f);
        }
        *(v8h*)(Hdst + pos * DST_STRIDE + chbase) = outh;
    }
}

// ---------------------------------------------------------------------------
// Gather + 3-layer MLP + sample-max.  One WG = 4 points x 32 samples.
// ---------------------------------------------------------------------------
__global__ __launch_bounds__(256) void mlp_kernel(
    const float* __restrict__ xyz, const float* __restrict__ points,
    const float* __restrict__ new_xyz_ws, const int* __restrict__ ball_idx,
    const _Float16* __restrict__ w0h, const _Float16* __restrict__ w1h,
    const _Float16* __restrict__ w2h, const float* __restrict__ ab,
    float* __restrict__ out)
{
    __shared__ __align__(16) _Float16 H1[128 * 72];    // mid buffer (stride 72)
    __shared__ __align__(16) _Float16 HX[128 * 104];   // H0 (stride 104) / H1b (stride 72)
    __shared__ float pool[4 * 128];

    const int t = threadIdx.x, lane = t & 31;
    const int wid = __builtin_amdgcn_readfirstlane(t >> 5);
    const int b = blockIdx.x >> 8;
    const int pt0 = (blockIdx.x & 255) * 4;
    const float* xb = xyz + (size_t)b * 3 * NPTS;
    const float* pb = points + (size_t)b * 64 * NPTS;

    for (int i = t; i < 512; i += 256) pool[i] = 0.f;

    // stage layer-0 activations: [pos=lp*32+k][ch]; ch 0..2 = xyz-centroid,
    // 3..66 = points, 67..95 zero pad.
    for (int i = t; i < 128 * 96; i += 256) {
        int pos = i / 96, ch = i % 96;
        int lp = pos >> 5, k = pos & 31;
        int g = (b << 10) + pt0 + lp;
        int j = ball_idx[g * 32 + k];
        float val;
        if (ch < 3)       val = xb[ch * NPTS + j] - new_xyz_ws[g * 3 + ch];
        else if (ch < 67) val = pb[(ch - 3) * NPTS + j];
        else              val = 0.f;
        HX[pos * 104 + ch] = (_Float16)val;
    }
    __syncthreads();

    // layer 0: 96 -> 64   (HX/104 -> H1/72)
    gemm_layer<96, 104, 72, 64, 96>(w0h, HX, H1, ab + 0, ab + 64, lane, wid);
    __syncthreads();
    // layer 1: 64 -> 64   (H1/72 -> HX reused with stride 72)
    gemm_layer<64, 72, 72, 64, 64>(w1h, H1, HX, ab + 128, ab + 192, lane, wid);
    __syncthreads();

    // layer 2: 64 -> 128, fused sample-max into pool
    {
        const float* a2  = ab + 256;
        const float* bb2 = ab + 384;
        const int hipart = (lane & 16) >> 1;
        const int pos = (wid << 4) + (lane & 15);
        const _Float16* hcol = HX + pos * 72 + hipart;

        v16h bfrag[2];
        #pragma unroll
        for (int kk = 0; kk < 2; ++kk) {
            v8h blo = *(const v8h*)(hcol + kk * 32);
            v8h bhi = *(const v8h*)(hcol + kk * 32 + 16);
            bfrag[kk] = cat8(blo, bhi);
        }

        #pragma unroll
        for (int j = 0; j < 8; ++j) {
            const int m0 = j << 4;
            v8f acc = {0.f, 0.f, 0.f, 0.f, 0.f, 0.f, 0.f, 0.f};
            const _Float16* wrow = w2h + (m0 + (lane & 15)) * 64 + hipart;
            #pragma unroll
            for (int kk = 0; kk < 2; ++kk) {
                v8h alo = *(const v8h*)(wrow + kk * 32);
                v8h ahi = *(const v8h*)(wrow + kk * 32 + 16);
                acc = __builtin_amdgcn_wmma_f32_16x16x32_f16(
                    false, cat8(alo, ahi), false, bfrag[kk], (short)0, acc, false, false);
            }
            float vals[8];
            const int chbase = m0 + hipart;
            #pragma unroll
            for (int r = 0; r < 8; ++r)
                vals[r] = fmaxf(acc[r] * a2[chbase + r] + bb2[chbase + r], 0.f);
            // max over the 16 position-lanes (16 samples of one point)
            #pragma unroll
            for (int off = 8; off >= 1; off >>= 1)
                #pragma unroll
                for (int r = 0; r < 8; ++r)
                    vals[r] = fmaxf(vals[r], __shfl_xor(vals[r], off, 32));
            if ((lane & 15) == 0) {
                int lp = wid >> 1;                    // n0 = wid*16 -> point
                #pragma unroll
                for (int r = 0; r < 8; ++r)
                    atomicMax((int*)&pool[lp * 128 + chbase + r],
                              __float_as_int(vals[r]));   // relu => >= 0, int-cmp ok
            }
        }
    }
    __syncthreads();

    for (int i = t; i < 512; i += 256) {
        int lp = i >> 7, ch = i & 127;
        out[49152 + ((size_t)(b * 128 + ch)) * NPOINT_ + (pt0 + lp)] = pool[lp * 128 + ch];
    }
}

// ---------------------------------------------------------------------------
extern "C" void kernel_launch(void* const* d_in, const int* in_sizes, int n_in,
                              void* d_out, int out_size, void* d_ws, size_t ws_size,
                              hipStream_t stream)
{
    (void)in_sizes; (void)n_in; (void)out_size; (void)ws_size;
    const float* xyz    = (const float*)d_in[0];
    const float* points = (const float*)d_in[1];
    const float* w0  = (const float*)d_in[2];
    const float* b0  = (const float*)d_in[3];
    const float* g0  = (const float*)d_in[4];
    const float* be0 = (const float*)d_in[5];
    const float* m0  = (const float*)d_in[6];
    const float* v0  = (const float*)d_in[7];
    const float* w1  = (const float*)d_in[8];
    const float* b1  = (const float*)d_in[9];
    const float* g1  = (const float*)d_in[10];
    const float* be1 = (const float*)d_in[11];
    const float* m1  = (const float*)d_in[12];
    const float* v1  = (const float*)d_in[13];
    const float* w2  = (const float*)d_in[14];
    const float* b2  = (const float*)d_in[15];
    const float* g2  = (const float*)d_in[16];
    const float* be2 = (const float*)d_in[17];
    const float* m2  = (const float*)d_in[18];
    const float* v2  = (const float*)d_in[19];
    float* out = (float*)d_out;

    char* ws = (char*)d_ws;
    int*      fps_idx    = (int*)ws;                       //  65,536 B
    float*    new_xyz_ws = (float*)(ws + 65536);           // 196,608 B
    int*      ball       = (int*)(ws + 262144);            // 2,097,152 B
    _Float16* w0h        = (_Float16*)(ws + 2359296);      //  12,288 B
    _Float16* w1h        = (_Float16*)(ws + 2371584);      //   8,192 B
    _Float16* w2h        = (_Float16*)(ws + 2379776);      //  16,384 B
    float*    ab         = (float*)(ws + 2396160);         //   2,048 B

    prep_kernel<<<1, 256, 0, stream>>>(w0, b0, g0, be0, m0, v0,
                                       w1, b1, g1, be1, m1, v1,
                                       w2, b2, g2, be2, m2, v2,
                                       w0h, w1h, w2h, ab);
    fps_kernel<<<16, 512, 0, stream>>>(xyz, fps_idx, new_xyz_ws, out);
    ball_kernel<<<2048, 256, 0, stream>>>(xyz, new_xyz_ws, ball);
    mlp_kernel<<<4096, 256, 0, stream>>>(xyz, points, new_xyz_ws, ball,
                                         w0h, w1h, w2h, ab, out);
}
